// CausalConv1d_22608707846991
// MI455X (gfx1250) — matrix operand order repaired and checked by reading, not compiled, not run on hardware
//
#include <hip/hip_runtime.h>

// ---------------------------------------------------------------------------
// Causal FFT convolution for MI455X (gfx1250, wave32).
//   out = irfft( rfft(pad_left(x, K-1)) * conj(rfft(pad_left(w,1), N)) )[:L] + bias
// B=32, Cin=Cout=128, L=K=4096, N=8192, F=N/2+1=4097 Hermitian bins.
// FFTs: in-LDS Stockham radix-4 (6x radix-4 + 1x radix-2, autosorting,
//       7 LDS round-trips instead of 13 for radix-2).
// Frequency-domain channel contraction runs on v_wmma_f32_16x16x32_f16.
// ---------------------------------------------------------------------------

typedef __attribute__((ext_vector_type(16))) _Float16 v16h;
typedef __attribute__((ext_vector_type(8)))  float    v8f;

#define B_SZ   32
#define CIN    128
#define COUT   128
#define L_SZ   4096
#define NFFT   8192
#define NF     4097                 // rfft bins stored (0..N/2)
#define SEQ_X  (B_SZ * CIN)         // 4096 x-sequences
#define SEQ_W  (COUT * CIN)         // 16384 weight-sequences
#define SEQ_O  (B_SZ * COUT)        // 4096 output sequences

// ---------------- shared in-LDS Stockham radix-4 FFT (length 8192) ----------

__device__ __forceinline__ void fill_tw(float2* tw) {
  // tw[j] = e^{-2*pi*i*j/N}, j = 0..N/2-1
  for (int j = threadIdx.x; j < NFFT / 2; j += blockDim.x) {
    float ang = -6.283185307179586f * (float)j / (float)NFFT;
    float s, c;
    __sincosf(ang, &s, &c);
    tw[j] = make_float2(c, s);
  }
}

__device__ __forceinline__ float2 cmul(float2 a, float2 b) {
  return make_float2(a.x * b.x - a.y * b.y, a.x * b.y + a.y * b.x);
}

// Input in bufA (natural order). Returns pointer to in-order result.
__device__ float2* fft8192_r4(float2* bufA, float2* bufB, const float2* tw) {
  float2* src = bufA;
  float2* dst = bufB;
  int n = NFFT;
  int log2s = 0;
  __syncthreads();
  // six radix-4 stages: n = 8192,2048,512,128,32,8
  while (n > 2) {
    const int m     = n >> 2;
    const int s     = 1 << log2s;
    const int tstep = NFFT / n;              // step for w1 in the N/2 table
    for (int t = threadIdx.x; t < NFFT / 4; t += blockDim.x) {
      const int q = t & (s - 1);
      const int p = t >> log2s;
      const float2 a = src[q + s * p];
      const float2 b = src[q + s * (p + m)];
      const float2 c = src[q + s * (p + 2 * m)];
      const float2 d = src[q + s * (p + 3 * m)];
      const float apc_x = a.x + c.x, apc_y = a.y + c.y;
      const float amc_x = a.x - c.x, amc_y = a.y - c.y;
      const float bpd_x = b.x + d.x, bpd_y = b.y + d.y;
      const float bmd_x = b.x - d.x, bmd_y = b.y - d.y;
      const float2 w1 = tw[p * tstep];           // idx < N/4
      const float2 w2 = tw[2 * p * tstep];       // idx < N/2
      const float2 w3 = cmul(w1, w2);            // avoid table idx >= N/2
      const int ob = q + s * 4 * p;
      // y0 = (a+c) + (b+d)
      dst[ob] = make_float2(apc_x + bpd_x, apc_y + bpd_y);
      // y1 = w1 * ((a-c) - j(b-d)) = w1 * (amc_x + bmd_y, amc_y - bmd_x)
      dst[ob + s]     = cmul(w1, make_float2(amc_x + bmd_y, amc_y - bmd_x));
      // y2 = w2 * ((a+c) - (b+d))
      dst[ob + 2 * s] = cmul(w2, make_float2(apc_x - bpd_x, apc_y - bpd_y));
      // y3 = w3 * ((a-c) + j(b-d)) = w3 * (amc_x - bmd_y, amc_y + bmd_x)
      dst[ob + 3 * s] = cmul(w3, make_float2(amc_x - bmd_y, amc_y + bmd_x));
    }
    __syncthreads();
    float2* tmp = src; src = dst; dst = tmp;
    n >>= 2;
    log2s += 2;
  }
  // final radix-2 stage: n == 2, s == NFFT/2
  {
    const int s = 1 << log2s;                    // 4096
    for (int q = threadIdx.x; q < s; q += blockDim.x) {
      const float2 a = src[q];
      const float2 b = src[q + s];
      dst[q]     = make_float2(a.x + b.x, a.y + b.y);
      dst[q + s] = make_float2(a.x - b.x, a.y - b.y);
    }
    __syncthreads();
  }
  return dst;
}

// ------------------------- kernel 1: FFT of padded x ------------------------
// xp[n] = 0 for n<4096, x[n-4096] otherwise (causal left pad by K-1=4096).
__global__ __launch_bounds__(256)
void fft_x_kernel(const float* __restrict__ x,
                  _Float16* __restrict__ Xr, _Float16* __restrict__ Xi) {
  extern __shared__ __align__(16) char smem[];
  float2* bufA = (float2*)smem;                // 64 KB
  float2* bufB = bufA + NFFT;                  // 64 KB
  float2* tw   = bufB + NFFT;                  // 32 KB
  const int seq = blockIdx.x;                  // b*128 + c
  const float* xs = x + (size_t)seq * L_SZ;
  for (int t = threadIdx.x; t < NFFT; t += blockDim.x) {
    const float v = (t >= L_SZ) ? xs[t - L_SZ] : 0.0f;
    bufA[t] = make_float2(v, 0.0f);
  }
  fill_tw(tw);
  float2* res = fft8192_r4(bufA, bufB, tw);
  _Float16* pr = Xr + (size_t)seq * NF;
  _Float16* pi = Xi + (size_t)seq * NF;
  for (int k = threadIdx.x; k < NF; k += blockDim.x) {
    pr[k] = (_Float16)res[k].x;
    pi[k] = (_Float16)res[k].y;
  }
}

// ------------------------ kernel 3: FFT of padded w -------------------------
// w'[0]=0, w'[1..4096]=w[0..4095] (left pad 1 to odd K), zero-extended to N.
__global__ __launch_bounds__(256)
void fft_w_kernel(const float* __restrict__ w,
                  _Float16* __restrict__ Vr, _Float16* __restrict__ Vi) {
  extern __shared__ __align__(16) char smem[];
  float2* bufA = (float2*)smem;
  float2* bufB = bufA + NFFT;
  float2* tw   = bufB + NFFT;
  const int seq = blockIdx.x;                  // o*128 + c
  const float* ws = w + (size_t)seq * L_SZ;
  for (int t = threadIdx.x; t < NFFT; t += blockDim.x) {
    const float v = (t >= 1 && t <= L_SZ) ? ws[t - 1] : 0.0f;
    bufA[t] = make_float2(v, 0.0f);
  }
  fill_tw(tw);
  float2* res = fft8192_r4(bufA, bufB, tw);
  _Float16* pr = Vr + (size_t)seq * NF;        // NOT conjugated; conj folded
  _Float16* pi = Vi + (size_t)seq * NF;        // into contraction signs
  for (int k = threadIdx.x; k < NF; k += blockDim.x) {
    pr[k] = (_Float16)res[k].x;
    pi[k] = (_Float16)res[k].y;
  }
}

// ------------- kernels 2/4: f16 transpose [S][F] -> [F][S] ------------------
__global__ __launch_bounds__(256)
void transpose_f16_kernel(const _Float16* __restrict__ inR,
                          const _Float16* __restrict__ inI,
                          _Float16* __restrict__ outR,
                          _Float16* __restrict__ outI, int S, int F) {
  __shared__ _Float16 tile[32][33];
  const _Float16* in  = blockIdx.z ? inI : inR;
  _Float16*       out = blockIdx.z ? outI : outR;
  const int f0 = blockIdx.x * 32;
  const int s0 = blockIdx.y * 32;
  const int tx = threadIdx.x, ty = threadIdx.y;              // (32, 8)
  if (f0 + tx < F) {
    for (int ii = 0; ii < 4; ++ii) {
      const int sl = ty * 4 + ii;                            // S is a mult of 32
      tile[sl][tx] = in[(size_t)(s0 + sl) * F + (f0 + tx)];
    }
  }
  __syncthreads();
  for (int ii = 0; ii < 4; ++ii) {
    const int fl = ty * 4 + ii;
    if (f0 + fl < F) out[(size_t)(f0 + fl) * S + (s0 + tx)] = tile[tx][fl];
  }
}

// --------- kernel 5: per-bin complex contraction on WMMA (the star) ---------
// Out[k](32x128) = X[k](32x128c) * conj(V[k])(128o x 128c)^T per bin k.
//   out_r = sum_c Xr*Vr + Xi*Vi      (one chained f32 accumulator)
//   out_i = sum_c Xi*Vr - sum_c Xr*Vi (two accumulators, VALU subtract at end;
//           f16 WMMA NEG[1:0] must be 0 per ISA 7.12, so no neg_a/neg_b)
__global__ __launch_bounds__(512)
void contract_wmma_kernel(const _Float16* __restrict__ XrT,
                          const _Float16* __restrict__ XiT,
                          const _Float16* __restrict__ VrT,
                          const _Float16* __restrict__ ViT,
                          float* __restrict__ Or, float* __restrict__ Oi) {
  extern __shared__ __align__(16) char smem[];
  _Float16* sXr = (_Float16*)smem;          // 4096 halfs  (8 KB)
  _Float16* sXi = sXr + SEQ_X;
  _Float16* sVr = sXi + SEQ_X;              // 16384 halfs (32 KB)
  _Float16* sVi = sVr + SEQ_W;
  const int k = blockIdx.x;

  { // cooperative vectorized stage-in (uint4 = 8 halfs)
    const uint4* gxr = (const uint4*)(XrT + (size_t)k * SEQ_X);
    const uint4* gxi = (const uint4*)(XiT + (size_t)k * SEQ_X);
    const uint4* gvr = (const uint4*)(VrT + (size_t)k * SEQ_W);
    const uint4* gvi = (const uint4*)(ViT + (size_t)k * SEQ_W);
    uint4* lxr = (uint4*)sXr; uint4* lxi = (uint4*)sXi;
    uint4* lvr = (uint4*)sVr; uint4* lvi = (uint4*)sVi;
    for (int i = threadIdx.x; i < SEQ_X / 8; i += blockDim.x) {
      lxr[i] = gxr[i];
      lxi[i] = gxi[i];
    }
    for (int i = threadIdx.x; i < SEQ_W / 8; i += blockDim.x) {
      lvr[i] = gvr[i];
      lvi[i] = gvi[i];
    }
  }
  __syncthreads();

  const int lane = threadIdx.x & 31;
  const int wv   = threadIdx.x >> 5;        // 16 waves
  const int mb   = wv & 1;                  // 2 b-tiles of 16
  const int no   = wv >> 1;                 // 8 o-tiles of 16
  const int r16  = lane & 15;
  const int hi   = lane >> 4;
  // A 16x32 f16 per-lane (ISA 7.12.2): K = 8*hi+0..7 and 16+8*hi+0..7, row=lane%16
  const int abase = (mb * 16 + r16) * CIN;
  // B 32x16 f16 per-lane: K = 8*hi+0..7 and 16+8*hi+0..7, col = lane%16
  const int bbase = (no * 16 + r16) * CIN;

  v8f acc_r  = {};
  v8f acc_i1 = {};
  v8f acc_i2 = {};
  union V16 { v16h v; uint4 q[2]; };

#pragma unroll
  for (int kc = 0; kc < CIN / 32; ++kc) {   // 4 K-chunks of 32
    V16 ar, ai, br, bi;
    const int ao = abase + kc * 32 + 8 * hi;
    ar.q[0] = *(const uint4*)(sXr + ao);
    ar.q[1] = *(const uint4*)(sXr + ao + 16);
    ai.q[0] = *(const uint4*)(sXi + ao);
    ai.q[1] = *(const uint4*)(sXi + ao + 16);
    const int bo = bbase + kc * 32 + 8 * hi;
    br.q[0] = *(const uint4*)(sVr + bo);
    br.q[1] = *(const uint4*)(sVr + bo + 16);
    bi.q[0] = *(const uint4*)(sVi + bo);
    bi.q[1] = *(const uint4*)(sVi + bo + 16);

    acc_r  = __builtin_amdgcn_wmma_f32_16x16x32_f16(false, ar.v, false, br.v,
                                                    (short)0, acc_r,  false, false);
    acc_r  = __builtin_amdgcn_wmma_f32_16x16x32_f16(false, ai.v, false, bi.v,
                                                    (short)0, acc_r,  false, false);
    acc_i1 = __builtin_amdgcn_wmma_f32_16x16x32_f16(false, ai.v, false, br.v,
                                                    (short)0, acc_i1, false, false);
    acc_i2 = __builtin_amdgcn_wmma_f32_16x16x32_f16(false, ar.v, false, bi.v,
                                                    (short)0, acc_i2, false, false);
  }

  // D layout: VGPR j -> M = j + 8*hi (+16*mb), N = lane%16 (+16*no)
  const size_t base = (size_t)k * (B_SZ * COUT);
  const int colN = no * 16 + r16;
#pragma unroll
  for (int j = 0; j < 8; ++j) {
    const int m = mb * 16 + hi * 8 + j;
    const size_t idx = base + (size_t)m * COUT + colN;
    Or[idx] = acc_r[j];
    Oi[idx] = acc_i1[j] - acc_i2[j];
  }
}

// ------ kernel 6: Hermitian mirror + transpose [k][bo]f32 -> [bo][k]c64 -----
__global__ __launch_bounds__(256)
void mirror_transpose_kernel(const float* __restrict__ Or,
                             const float* __restrict__ Oi,
                             float2* __restrict__ Y) {
  __shared__ float2 tile[32][33];
  const int k0 = blockIdx.x * 32;
  const int s0 = blockIdx.y * 32;
  const int tx = threadIdx.x, ty = threadIdx.y;              // (32, 8)
  for (int ii = 0; ii < 4; ++ii) {
    const int i = ty * 4 + ii;
    const int k = k0 + i;
    const int  kp  = (k <= NFFT / 2) ? k : NFFT - k;         // conj symmetry
    const float sg = (k <= NFFT / 2) ? 1.0f : -1.0f;
    const size_t src = (size_t)kp * SEQ_O + (s0 + tx);
    tile[i][tx] = make_float2(Or[src], sg * Oi[src]);
  }
  __syncthreads();
  for (int ii = 0; ii < 4; ++ii) {
    const int i = ty * 4 + ii;
    Y[(size_t)(s0 + i) * NFFT + (k0 + tx)] = tile[tx][i];
  }
}

// ----------------- kernel 7: inverse FFT + crop + bias ----------------------
// real(IFFT(Y)) = real(FFT(conj(Y)))/N
__global__ __launch_bounds__(256)
void ifft_kernel(const float2* __restrict__ Y, const float* __restrict__ bias,
                 float* __restrict__ out) {
  extern __shared__ __align__(16) char smem[];
  float2* bufA = (float2*)smem;
  float2* bufB = bufA + NFFT;
  float2* tw   = bufB + NFFT;
  const int seq = blockIdx.x;                                // b*128 + o
  const float2* ys = Y + (size_t)seq * NFFT;
  for (int t = threadIdx.x; t < NFFT; t += blockDim.x) {
    const float2 v = ys[t];
    bufA[t] = make_float2(v.x, -v.y);                        // conjugate
  }
  fill_tw(tw);
  float2* res = fft8192_r4(bufA, bufB, tw);
  const float b   = bias[seq & (COUT - 1)];
  const float inv = 1.0f / (float)NFFT;
  float* os = out + (size_t)seq * L_SZ;
  for (int t = threadIdx.x; t < L_SZ; t += blockDim.x) {
    os[t] = res[t].x * inv + b;
  }
}

// ---------------------------------------------------------------------------
extern "C" void kernel_launch(void* const* d_in, const int* in_sizes, int n_in,
                              void* d_out, int out_size, void* d_ws, size_t ws_size,
                              hipStream_t stream) {
  (void)in_sizes; (void)n_in; (void)out_size; (void)ws_size;
  const float* x    = (const float*)d_in[0];   // [32][128][4096]
  const float* w    = (const float*)d_in[1];   // [128][128][4096]
  const float* bias = (const float*)d_in[2];   // [128]
  float* out = (float*)d_out;                  // [32][128][4096]

  // ---- workspace carve (sizes are 256B multiples; total ~671 MB) ----
  char* p = (char*)d_ws;
  const size_t szX = (size_t)SEQ_X * NF * sizeof(_Float16);  // 33.56 MB
  const size_t szW = (size_t)SEQ_W * NF * sizeof(_Float16);  // 134.25 MB
  _Float16* Xr  = (_Float16*)p; p += szX;
  _Float16* Xi  = (_Float16*)p; p += szX;
  _Float16* XrT = (_Float16*)p; p += szX;
  _Float16* XiT = (_Float16*)p; p += szX;
  _Float16* Vr  = (_Float16*)p; p += szW;
  _Float16* Vi  = (_Float16*)p; p += szW;
  _Float16* VrT = (_Float16*)p; p += szW;
  _Float16* ViT = (_Float16*)p; p += szW;
  // overlays (dead regions by the time they are written):
  float*  Or = (float*)Vr;    // 67.1 MB into Vr's 134 MB (Vr dead after xpose)
  float*  Oi = (float*)Vi;
  float2* Y  = (float2*)VrT;  // 268.4 MB spanning VrT+ViT (dead after contract)

  const size_t fftLds = 2 * NFFT * sizeof(float2)            // ping + pong
                      + (NFFT / 2) * sizeof(float2);         // twiddles: 160 KB
  const size_t conLds = (size_t)(2 * SEQ_X + 2 * SEQ_W) * sizeof(_Float16); // 80 KB
  const dim3 tb(32, 8);

  // 1) spectra of padded x
  fft_x_kernel<<<SEQ_X, 256, fftLds, stream>>>(x, Xr, Xi);
  // 2) [bc][k] -> [k][bc]
  transpose_f16_kernel<<<dim3((NF + 31) / 32, SEQ_X / 32, 2), tb, 0, stream>>>(
      Xr, Xi, XrT, XiT, SEQ_X, NF);
  // 3) spectra of padded w
  fft_w_kernel<<<SEQ_W, 256, fftLds, stream>>>(w, Vr, Vi);
  // 4) [oc][k] -> [k][oc]
  transpose_f16_kernel<<<dim3((NF + 31) / 32, SEQ_W / 32, 2), tb, 0, stream>>>(
      Vr, Vi, VrT, ViT, SEQ_W, NF);
  // 5) per-bin complex GEMM on WMMA (writes overlay region C)
  contract_wmma_kernel<<<NF, 512, conLds, stream>>>(XrT, XiT, VrT, ViT, Or, Oi);
  // 6) Hermitian mirror + transpose to [bo][8192] (writes overlay region D)
  mirror_transpose_kernel<<<dim3(NFFT / 32, SEQ_O / 32), tb, 0, stream>>>(Or, Oi, Y);
  // 7) inverse FFT, crop to L, add bias
  ifft_kernel<<<SEQ_O, 256, fftLds, stream>>>(Y, bias, out);
}
